// IALSHSelfAttention_46299747451200
// MI455X (gfx1250) — matrix operand order repaired and checked by reading, not compiled, not run on hardware
//
#include <hip/hip_runtime.h>

#define BATCH 2
#define SEQ   2048
#define DIMSZ 1024
#define HEADS 16
#define DH    64
#define NEGV  (-1.0e4f)
#define SCALE 0.125f   // 1/sqrt(64)

typedef __attribute__((ext_vector_type(16))) __bf16 v16bf;
typedef __attribute__((ext_vector_type(8)))  __bf16 v8bf;
typedef __attribute__((ext_vector_type(8)))  float  v8f;

static __device__ __forceinline__ v8f zero8() {
  v8f z;
#pragma unroll
  for (int i = 0; i < 8; ++i) z[i] = 0.0f;
  return z;
}

// A-fragment (16x32, M x K), 16-bit: lane holds row (lane&15);
// K chunks {g..g+7} and {16+g..16+g+7}, g = (lane>=16)*8.
static __device__ __forceinline__ v16bf load_a_f32(const float* base, int stride, int k0, int lane) {
  const int r = lane & 15;
  const int g = (lane >> 4) << 3;
  const float* p0 = base + (size_t)r * stride + k0 + g;
  const float* p1 = p0 + 16;
  v16bf f;
#pragma unroll
  for (int i = 0; i < 8; ++i) { f[i] = (__bf16)p0[i]; f[8 + i] = (__bf16)p1[i]; }
  return f;
}

// B-fragment (32x16, K x N), 16-bit: lane holds column (lane&15) of B
// (= row of B^T), K chunks {g..g+7} and {16+g..16+g+7}, g = (lane>=16)*8.
static __device__ __forceinline__ v16bf load_b_f32(const float* base, int stride, int k0, int lane) {
  const int nn = lane & 15;
  const int hb = (lane >> 4) << 3;
  const float* p = base + (size_t)nn * stride + k0 + hb;
  v16bf f;
#pragma unroll
  for (int i = 0; i < 8; ++i) { f[i] = (__bf16)p[i]; f[8 + i] = (__bf16)p[16 + i]; }
  return f;
}

static __device__ __forceinline__ v16bf load_a_bf16(const __bf16* base, int stride, int k0, int lane) {
  const int r = lane & 15;
  const int g = (lane >> 4) << 3;
  const __bf16* p0 = base + (size_t)r * stride + k0 + g;
  const __bf16* p1 = p0 + 16;
  v16bf f;
#pragma unroll
  for (int i = 0; i < 8; ++i) { f[i] = p0[i]; f[8 + i] = p1[i]; }
  return f;
}

static __device__ __forceinline__ v16bf load_b_bf16(const __bf16* base, int stride, int k0, int lane) {
  const int nn = lane & 15;
  const int hb = (lane >> 4) << 3;
  const __bf16* p = base + (size_t)nn * stride + k0 + hb;
  v16bf f;
#pragma unroll
  for (int i = 0; i < 8; ++i) { f[i] = p[i]; f[8 + i] = p[16 + i]; }
  return f;
}

// Async global->LDS 16B copy (gfx1250 VGLOBAL async op, ASYNCcnt-tracked).
// Inline asm is portable across both toolchains' builtin arities.
static __device__ __forceinline__ void async_copy_b128(unsigned lds_off, const void* gptr) {
  asm volatile("global_load_async_to_lds_b128 %0, %1, off"
               :: "v"(lds_off), "v"((unsigned long long)(size_t)gptr)
               : "memory");
}
static __device__ __forceinline__ void wait_async0() {
  asm volatile("s_wait_asynccnt 0" ::: "memory");
}

// ---------------------------------------------------------------------------
// Kernel 1: fused QK/V projection.  out[i,j] = sum_k hs[i,k]*W[j,k] + b[j]
// Each wave computes a 16x32 strip of BOTH qk and v (one shared A-fragment,
// 4 WMMAs per K-step).  qk_out: [B][H][L][DH] bf16 ; vT_out: [B][H][DH][L].
// ---------------------------------------------------------------------------
__global__ __launch_bounds__(256) void proj_kernel(
    const float* __restrict__ hs, const float* __restrict__ qk_w,
    const float* __restrict__ qk_b, const float* __restrict__ v_w,
    const float* __restrict__ v_b, __bf16* __restrict__ qk_out,
    __bf16* __restrict__ vT_out) {
  const int lane = threadIdx.x & 31;
  const int wave = threadIdx.x >> 5;
  const int tile = blockIdx.x * 8 + wave;       // 8192 wave-strips
  const int ntg  = tile & 31;                   // 32 N strips (DIM/32)
  const int mt   = tile >> 5;                   // 256 M tiles (B*L/16)
  const int m0 = mt * 16;
  const int n0 = ntg * 32;
  const int nn = lane & 15;

  v8f cq0, cq1, cv0, cv1;
  {
    const float b0 = qk_b[n0 + nn],      b1 = qk_b[n0 + 16 + nn];
    const float c0 = v_b[n0 + nn],       c1 = v_b[n0 + 16 + nn];
#pragma unroll
    for (int r = 0; r < 8; ++r) { cq0[r] = b0; cq1[r] = b1; cv0[r] = c0; cv1[r] = c1; }
  }

  const float* A   = hs   + (size_t)m0 * DIMSZ;
  const float* Bq0 = qk_w + (size_t)n0 * DIMSZ;
  const float* Bq1 = qk_w + (size_t)(n0 + 16) * DIMSZ;
  const float* Bv0 = v_w  + (size_t)n0 * DIMSZ;
  const float* Bv1 = v_w  + (size_t)(n0 + 16) * DIMSZ;

  for (int k0 = 0; k0 < DIMSZ; k0 += 32) {
    const v16bf a = load_a_f32(A, DIMSZ, k0, lane);
    v16bf bq = load_b_f32(Bq0, DIMSZ, k0, lane);
    cq0 = __builtin_amdgcn_wmma_f32_16x16x32_bf16(false, a, false, bq, (short)0, cq0, false, false);
    bq = load_b_f32(Bq1, DIMSZ, k0, lane);
    cq1 = __builtin_amdgcn_wmma_f32_16x16x32_bf16(false, a, false, bq, (short)0, cq1, false, false);
    bq = load_b_f32(Bv0, DIMSZ, k0, lane);
    cv0 = __builtin_amdgcn_wmma_f32_16x16x32_bf16(false, a, false, bq, (short)0, cv0, false, false);
    bq = load_b_f32(Bv1, DIMSZ, k0, lane);
    cv1 = __builtin_amdgcn_wmma_f32_16x16x32_bf16(false, a, false, bq, (short)0, cv1, false, false);
  }

  const int b   = m0 >> 11;          // / SEQ
  const int l0  = m0 & (SEQ - 1);
  const int hh  = n0 >> 6;           // head (n0 % 64 in {0,32} => same head)
  const int dh0 = n0 & (DH - 1);
  const int mb  = (lane < 16) ? 0 : 8;

  // qk[b][hh][l0+mb+r][dh] : strided 2B stores (two 16-col subtiles)
  __bf16* q = qk_out + (((size_t)b * HEADS + hh) * SEQ + (l0 + mb)) * DH + dh0 + nn;
#pragma unroll
  for (int r = 0; r < 8; ++r) {
    q[(size_t)r * DH]      = (__bf16)cq0[r];
    q[(size_t)r * DH + 16] = (__bf16)cq1[r];
  }

  // vT[b][hh][dh][l0+mb .. +8] : contiguous 16B stores
  v8bf p0, p1;
#pragma unroll
  for (int r = 0; r < 8; ++r) { p0[r] = (__bf16)cv0[r]; p1[r] = (__bf16)cv1[r]; }
  __bf16* vp0 = vT_out + (((size_t)b * HEADS + hh) * DH + (dh0 + nn)) * SEQ + l0 + mb;
  __bf16* vp1 = vp0 + (size_t)16 * SEQ;
  *reinterpret_cast<v8bf*>(vp0) = p0;
  *reinterpret_cast<v8bf*>(vp1) = p1;
}

// ---------------------------------------------------------------------------
// Kernel 2: IALSH hash bits.  One block per (b,h).
// hq = sign(qk . a[:64]) ; hk = sign(s*dot + (.5-nk2)*a64 + (.5-nk2^2)*a65)
// ---------------------------------------------------------------------------
__global__ __launch_bounds__(256) void hash_kernel(
    const __bf16* __restrict__ qk, const float* __restrict__ hash_a,
    unsigned* __restrict__ hqb, unsigned* __restrict__ hkb) {
  __shared__ float sa[DH + 2];
  __shared__ float red[256];
  const int bh = blockIdx.x;
  const __bf16* base = qk + (size_t)bh * SEQ * DH;
  if (threadIdx.x < DH + 2) sa[threadIdx.x] = hash_a[threadIdx.x];
  __syncthreads();

  float dots[8], n2s[8];
  float mymax = 0.0f;
#pragma unroll
  for (int it = 0; it < 8; ++it) {
    const int l = it * 256 + threadIdx.x;
    const __bf16* row = base + (size_t)l * DH;
    float dot = 0.0f, n2 = 0.0f;
#pragma unroll
    for (int d = 0; d < DH; ++d) {
      const float x = (float)row[d];
      dot += x * sa[d];
      n2  += x * x;
    }
    dots[it] = dot; n2s[it] = n2;
    mymax = fmaxf(mymax, n2);
  }
  red[threadIdx.x] = mymax;
  __syncthreads();
  for (int s2 = 128; s2 > 0; s2 >>= 1) {
    if (threadIdx.x < (unsigned)s2) red[threadIdx.x] = fmaxf(red[threadIdx.x], red[threadIdx.x + s2]);
    __syncthreads();
  }
  const float maxnorm = sqrtf(red[0]);
  const float s = 0.75f / fmaxf(maxnorm, 1e-12f);
  const float a64 = sa[DH], a65 = sa[DH + 1];

#pragma unroll
  for (int it = 0; it < 8; ++it) {
    const int l = it * 256 + threadIdx.x;
    const float nk2 = n2s[it] * s * s;
    const float hkv = s * dots[it] + (0.5f - nk2) * a64 + (0.5f - nk2 * nk2) * a65;
    const unsigned bq = __builtin_amdgcn_ballot_w32(dots[it] >= 0.0f);
    const unsigned bk = __builtin_amdgcn_ballot_w32(hkv >= 0.0f);
    if ((threadIdx.x & 31) == 0) {
      hqb[bh * (SEQ / 32) + (l >> 5)] = bq;
      hkb[bh * (SEQ / 32) + (l >> 5)] = bk;
    }
  }
}

// ---------------------------------------------------------------------------
// Kernel 3: flash attention.  grid = B*H*(L/128); block = 256 (8 waves);
// wave owns a 16-row query tile.  K/V tiles (shared by all 8 waves) are
// staged once per block into LDS with async global->LDS copies.
// ---------------------------------------------------------------------------
__global__ __launch_bounds__(256) void attn_kernel(
    const __bf16* __restrict__ qk, const __bf16* __restrict__ vT,
    const unsigned* __restrict__ hqb, const unsigned* __restrict__ hkb,
    float* __restrict__ out) {
  __shared__ float  sS[8][16 * 32];
  __shared__ __bf16 __attribute__((aligned(16))) sP[8][16 * 32];
  __shared__ float  sC[8][16];
  __shared__ __bf16 __attribute__((aligned(16))) sK[32 * DH];   // [key][dh]
  __shared__ __bf16 __attribute__((aligned(16))) sV[DH * 32];   // [dh][key]

  const unsigned t = threadIdx.x;
  const int lane = t & 31;
  const int wave = t >> 5;
  const int bh   = blockIdx.x >> 4;
  const int qblk = blockIdx.x & 15;
  const int q0   = qblk * 128 + wave * 16;
  const int b    = bh >> 4;
  const int h    = bh & 15;
  const int nn   = lane & 15;
  const int mb   = (lane < 16) ? 0 : 8;

  const __bf16* kv = qk + (size_t)bh * SEQ * DH;   // [l][dh]
  const __bf16* vt = vT + (size_t)bh * DH * SEQ;   // [dh][l]

  const v16bf qa0 = load_a_bf16(kv + (size_t)q0 * DH, DH, 0,  lane);
  const v16bf qa1 = load_a_bf16(kv + (size_t)q0 * DH, DH, 32, lane);

  const unsigned hqw  = hqb[bh * (SEQ / 32) + (q0 >> 5)];
  const unsigned hq16 = (hqw >> (q0 & 31)) & 0xFFFFu;

  // LDS offsets for async copies (low 32 bits of generic ptr = LDS address)
  const unsigned sK_off = (unsigned)(size_t)(&sK[0]) + t * 16u;
  const int vr = t >> 2;                   // 0..63 : dh row of V tile
  const int vc = (t & 3) << 3;             // 0,8,16,24 : key chunk (elements)
  const unsigned sV_off = (unsigned)(size_t)(&sV[0]) + (unsigned)(vr * 32 + vc) * 2u;

  v8f o0 = zero8(), o1 = zero8(), o2 = zero8(), o3 = zero8();
  float run_max = -3.0e38f, run_sum = 0.0f;

  for (int k0 = 0; k0 < SEQ; k0 += 32) {
    // ---- stage K tile (4KB contiguous) + V tile (64 rows x 64B) in LDS ----
    const char* gk = (const char*)(kv + (size_t)k0 * DH) + t * 16u;
    async_copy_b128(sK_off, gk);
    const __bf16* gv = vt + (size_t)vr * SEQ + k0 + vc;
    async_copy_b128(sV_off, gv);
    if (k0 + 32 < SEQ) {                    // prefetch next tiles into cache
      __builtin_prefetch(gk + 32 * DH * sizeof(__bf16), 0, 3);
      __builtin_prefetch(gv + 32, 0, 3);
    }
    wait_async0();
    __syncthreads();

    const unsigned hkw = hkb[bh * (SEQ / 32) + (k0 >> 5)];

    // ---- S = Q K^T (two 16-key subtiles), scale + mask, spill to LDS ----
#pragma unroll
    for (int sct = 0; sct < 2; ++sct) {
      v16bf b0 = load_b_bf16(&sK[sct * 16 * DH], DH, 0,  lane);
      v16bf b1 = load_b_bf16(&sK[sct * 16 * DH], DH, 32, lane);
      v8f sc = zero8();
      sc = __builtin_amdgcn_wmma_f32_16x16x32_bf16(false, qa0, false, b0, (short)0, sc, false, false);
      sc = __builtin_amdgcn_wmma_f32_16x16x32_bf16(false, qa1, false, b1, (short)0, sc, false, false);
      const unsigned kbit = (hkw >> (sct * 16 + nn)) & 1u;
#pragma unroll
      for (int r = 0; r < 8; ++r) {
        const int m = mb + r;
        const float mask = (((hq16 >> m) ^ kbit) & 1u) ? NEGV : 0.0f;
        sS[wave][m * 32 + sct * 16 + nn] = sc[r] * SCALE + mask;
      }
    }
    __syncthreads();

    // ---- online softmax: lanes 0..15 each own one query row ----
    if (lane < 16) {
      const int m = lane;
      float rowmax = -3.0e38f;
#pragma unroll
      for (int jj = 0; jj < 32; ++jj) rowmax = fmaxf(rowmax, sS[wave][m * 32 + jj]);
      const float nm   = fmaxf(run_max, rowmax);
      const float corr = __expf(run_max - nm);
      float rs = 0.0f;
#pragma unroll
      for (int jj = 0; jj < 32; ++jj) {
        const float p = __expf(sS[wave][m * 32 + jj] - nm);
        rs += p;
        sP[wave][m * 32 + jj] = (__bf16)p;
      }
      run_sum = run_sum * corr + rs;
      run_max = nm;
      sC[wave][m] = corr;
    }
    __syncthreads();

    // ---- rescale O and accumulate P * V (dh = 4 x 16 columns) ----
    float cf[8];
#pragma unroll
    for (int r = 0; r < 8; ++r) cf[r] = sC[wave][mb + r];
#pragma unroll
    for (int r = 0; r < 8; ++r) {
      o0[r] *= cf[r]; o1[r] *= cf[r]; o2[r] *= cf[r]; o3[r] *= cf[r];
    }
    const v16bf pa = load_a_bf16(&sP[wave][0], 32, 0, lane);
    v16bf vb0 = load_b_bf16(&sV[0 * 16 * 32], 32, 0, lane);
    v16bf vb1 = load_b_bf16(&sV[1 * 16 * 32], 32, 0, lane);
    v16bf vb2 = load_b_bf16(&sV[2 * 16 * 32], 32, 0, lane);
    v16bf vb3 = load_b_bf16(&sV[3 * 16 * 32], 32, 0, lane);
    o0 = __builtin_amdgcn_wmma_f32_16x16x32_bf16(false, pa, false, vb0, (short)0, o0, false, false);
    o1 = __builtin_amdgcn_wmma_f32_16x16x32_bf16(false, pa, false, vb1, (short)0, o1, false, false);
    o2 = __builtin_amdgcn_wmma_f32_16x16x32_bf16(false, pa, false, vb2, (short)0, o2, false, false);
    o3 = __builtin_amdgcn_wmma_f32_16x16x32_bf16(false, pa, false, vb3, (short)0, o3, false, false);
    __syncthreads();
  }

  // ---- epilogue: divide by row sums, store f32 output [b][l][h*64+dh] ----
  if (lane < 16) sC[wave][lane] = 1.0f / run_sum;
  __syncthreads();
  float inv[8];
#pragma unroll
  for (int r = 0; r < 8; ++r) inv[r] = sC[wave][mb + r];
  float* ob = out + ((size_t)b * SEQ + (q0 + mb)) * DIMSZ + h * DH;
#pragma unroll
  for (int r = 0; r < 8; ++r) {
    ob[(size_t)r * DIMSZ +  0 + nn] = o0[r] * inv[r];
    ob[(size_t)r * DIMSZ + 16 + nn] = o1[r] * inv[r];
    ob[(size_t)r * DIMSZ + 32 + nn] = o2[r] * inv[r];
    ob[(size_t)r * DIMSZ + 48 + nn] = o3[r] * inv[r];
  }
}

// ---------------------------------------------------------------------------
extern "C" void kernel_launch(void* const* d_in, const int* in_sizes, int n_in,
                              void* d_out, int out_size, void* d_ws, size_t ws_size,
                              hipStream_t stream) {
  const float* hs   = (const float*)d_in[0];
  const float* qk_w = (const float*)d_in[1];
  const float* qk_b = (const float*)d_in[2];
  const float* v_w  = (const float*)d_in[3];
  const float* v_b  = (const float*)d_in[4];
  const float* ha   = (const float*)d_in[5];
  float* out = (float*)d_out;

  char* ws = (char*)d_ws;
  __bf16*   qk  = (__bf16*)ws;                                  // 8 MiB
  __bf16*   vT  = (__bf16*)(ws + ((size_t)8 << 20));            // 8 MiB
  unsigned* hqb = (unsigned*)(ws + ((size_t)16 << 20));         // 8 KiB
  unsigned* hkb = hqb + BATCH * HEADS * (SEQ / 32);             // 8 KiB

  // 8192 wave-strips (16x32 x {qk,v}) over the 4096x1024 projection
  proj_kernel<<<1024, 256, 0, stream>>>(hs, qk_w, qk_b, v_w, v_b, qk, vT);
  hash_kernel<<<BATCH * HEADS, 256, 0, stream>>>(qk, ha, hqb, hkb);
  attn_kernel<<<BATCH * HEADS * (SEQ / 128), 256, 0, stream>>>(qk, vT, hqb, hkb, out);
}
